// DeepseekV4Indexer_6287832121446
// MI455X (gfx1250) — compile-verified, matrix operand order checked by reading, not run on hardware
//
#include <hip/hip_runtime.h>

// ---------------------------------------------------------------------------
// DeepseekV4 indexer for gfx1250 (MI455X), wave32 + WMMA bf16 + async LDS.
// ---------------------------------------------------------------------------

typedef __attribute__((ext_vector_type(16))) __bf16 v16bf;
typedef __attribute__((ext_vector_type(8)))  __bf16 v8bf;
typedef __attribute__((ext_vector_type(8)))  float  v8f;
typedef int v4i_vec __attribute__((vector_size(16)));  // async-LDS builtin operand

#define T_TOK   4096
#define H_SIZE  7168
#define QLR     1536
#define NHEADS  64
#define HDIM    128
#define QCOLS   (NHEADS * HDIM)   // 8192
#define TOPK    2048
#define KW_N    (HDIM + NHEADS)   // 192
#define Q_SCALE 0.08838834764831845f   // 128^-0.5
#define W_SCALE 0.125f                 // 64^-0.5
#define NEG_LOG_BASE_OVER_HALF (-0.28782313662425576f) // -ln(10000)/32

// ---- workspace layout (bytes) ---------------------------------------------
#define OFF_QRB   ((size_t)0)            // qr bf16       12,582,912
#define OFF_WQBT  ((size_t)12582912)     // wq_b^T bf16   25,165,824
#define OFF_HIDB  ((size_t)37748736)     // hidden bf16   58,720,256 (dead after gemm_kw)
#define OFF_WKT   ((size_t)96468992)     // [wk|wproj]^T   2,752,512 (dead after gemm_kw)
#define OFF_KWR   ((size_t)99221504)     // kw raw f32     3,145,728 (dead after ln_rope)
#define OFF_QB    ((size_t)37748736)     // q bf16 (rope'd) 67,108,864  overlays HIDB..KWR
#define OFF_KB    ((size_t)104857600)    // k bf16         1,048,576
#define OFF_W     ((size_t)105906176)    // w f32          1,048,576
// total ws needed: 106,954,752 bytes

// ---- CDNA5 async global->LDS path (guarded; falls back to load+ds_store) --
#if defined(__HIP_DEVICE_COMPILE__) && \
    __has_builtin(__builtin_amdgcn_global_load_async_to_lds_b128) && \
    __has_builtin(__builtin_amdgcn_s_wait_asynccnt)
#define USE_ASYNC_LDS 1
#else
#define USE_ASYNC_LDS 0
#endif

// ---------------------------------------------------------------------------
// WMMA helpers (CDNA5 layouts, wave32)
// A tile 16x32 bf16 from row-major MxK: lane L -> row L%16, kbase = (L&16)?8:0,
// elements [0..7] = K kbase..kbase+7, [8..15] = K kbase+16..kbase+23 -> 2x b128.
// B tile 32x16 bf16 from row-major NxK (column n = lane L%16): same pattern.
// ---------------------------------------------------------------------------
__device__ __forceinline__ v16bf load_tile16x32(const __bf16* base, int ld,
                                                int row0, int k0, int lane) {
  const int r  = row0 + (lane & 15);
  const int kb = (lane & 16) ? 8 : 0;
  const __bf16* p = base + (size_t)r * ld + k0 + kb;
  v8bf lo = *(const v8bf*)(p);
  v8bf hi = *(const v8bf*)(p + 16);
  v16bf out;
#pragma unroll
  for (int j = 0; j < 8; ++j) { out[j] = lo[j]; out[j + 8] = hi[j]; }
  return out;
}

// same tile pattern, but base points at a 16xHDIM strip in LDS -> ds_load_b128
__device__ __forceinline__ v16bf load_tile_lds(const __bf16* base, int k0, int lane) {
  const int r  = lane & 15;
  const int kb = (lane & 16) ? 8 : 0;
  const __bf16* p = base + r * HDIM + k0 + kb;
  v8bf lo = *(const v8bf*)(p);
  v8bf hi = *(const v8bf*)(p + 16);
  v16bf out;
#pragma unroll
  for (int j = 0; j < 8; ++j) { out[j] = lo[j]; out[j + 8] = hi[j]; }
  return out;
}

__device__ __forceinline__ v8f wmma_bf16(v16bf a, v16bf b, v8f c) {
  return __builtin_amdgcn_wmma_f32_16x16x32_bf16(false, a, false, b,
                                                 (short)0, c, false, false);
}

__device__ __forceinline__ v8f v8f_zero() {
  v8f z = {0.f, 0.f, 0.f, 0.f, 0.f, 0.f, 0.f, 0.f};
  return z;
}

// ---------------------------------------------------------------------------
// elementwise f32 -> bf16
// ---------------------------------------------------------------------------
__global__ void cvt_f32_bf16(const float* __restrict__ in,
                             __bf16* __restrict__ out, int n) {
  int i = blockIdx.x * blockDim.x + threadIdx.x;
  if (i < n) out[i] = (__bf16)in[i];
}

// ---------------------------------------------------------------------------
// tiled transpose + cvt: in f32 (RxC) -> out bf16 out[c*outStride + r]
// ---------------------------------------------------------------------------
__global__ void transpose_cvt(const float* __restrict__ in,
                              __bf16* __restrict__ out,
                              int R, int C, int outStride) {
  __shared__ float tile[32][33];
  int c0 = blockIdx.x * 32, r0 = blockIdx.y * 32;
  int tx = threadIdx.x, ty = threadIdx.y;
#pragma unroll
  for (int i = 0; i < 4; ++i) {
    int r = r0 + ty + i * 8, c = c0 + tx;
    if (r < R && c < C) tile[ty + i * 8][tx] = in[(size_t)r * C + c];
  }
  __syncthreads();
#pragma unroll
  for (int i = 0; i < 4; ++i) {
    int c = c0 + ty + i * 8, r = r0 + tx;
    if (c < C && r < R) out[(size_t)c * outStride + r] = (__bf16)tile[tx][ty + i * 8];
  }
}

// ---------------------------------------------------------------------------
// kw GEMM: hidden_bf16(4096x7168) @ [wk|wproj]^T(192x7168)^T -> kw_raw f32
// one wave per 16x64 tile
// ---------------------------------------------------------------------------
__global__ __launch_bounds__(32) void gemm_kw_kernel(
    const __bf16* __restrict__ A, const __bf16* __restrict__ BT,
    float* __restrict__ out) {
  const int lane = threadIdx.x;
  const int n0 = blockIdx.x * 64, i0 = blockIdx.y * 16;
  v8f acc[4];
#pragma unroll
  for (int t = 0; t < 4; ++t) acc[t] = v8f_zero();

  for (int k0 = 0; k0 < H_SIZE; k0 += 32) {
    v16bf a = load_tile16x32(A, H_SIZE, i0, k0, lane);
#pragma unroll
    for (int t = 0; t < 4; ++t) {
      v16bf b = load_tile16x32(BT, H_SIZE, n0 + t * 16, k0, lane);
      acc[t] = wmma_bf16(a, b, acc[t]);
    }
  }
  const int half8 = (lane & 16) ? 8 : 0;
  const int ln = lane & 15;
#pragma unroll
  for (int t = 0; t < 4; ++t)
#pragma unroll
    for (int v = 0; v < 8; ++v)
      out[(size_t)(i0 + v + half8) * KW_N + n0 + t * 16 + ln] = acc[t][v];
}

// ---------------------------------------------------------------------------
// per-row layernorm + rope on k, scale on w. block = 128 threads, grid = T.
// ---------------------------------------------------------------------------
__global__ __launch_bounds__(128) void ln_rope_kw(
    const float* __restrict__ kwraw, const float* __restrict__ gamma,
    const float* __restrict__ beta, const int* __restrict__ pos,
    __bf16* __restrict__ kout, float* __restrict__ wout) {
  __shared__ float red[128];
  __shared__ float kk[128];
  __shared__ float mu_s, var_s;
  const int row = blockIdx.x, d = threadIdx.x;

  float kv = kwraw[(size_t)row * KW_N + d];
  red[d] = kv; __syncthreads();
#pragma unroll
  for (int s = 64; s > 0; s >>= 1) { if (d < s) red[d] += red[d + s]; __syncthreads(); }
  if (d == 0) mu_s = red[0] * (1.f / 128.f);
  __syncthreads();
  const float mu = mu_s;
  const float dv = kv - mu;
  red[d] = dv * dv; __syncthreads();
#pragma unroll
  for (int s = 64; s > 0; s >>= 1) { if (d < s) red[d] += red[d + s]; __syncthreads(); }
  if (d == 0) var_s = red[0] * (1.f / 128.f);
  __syncthreads();

  kk[d] = dv * rsqrtf(var_s + 1e-6f) * gamma[d] + beta[d];
  __syncthreads();

  const int p = pos[row];
  if (d < 32) {
    float x1 = kk[d], x2 = kk[d + 32];
    float inv = __expf((float)d * NEG_LOG_BASE_OVER_HALF);
    float ang = (float)p * inv;
    float s_ = __sinf(ang), c_ = __cosf(ang);
    kout[(size_t)row * HDIM + d]      = (__bf16)(x1 * c_ - x2 * s_);
    kout[(size_t)row * HDIM + d + 32] = (__bf16)(x2 * c_ + x1 * s_);
  } else if (d >= 64) {
    kout[(size_t)row * HDIM + d] = (__bf16)kk[d];
  }
  if (d < NHEADS)
    wout[(size_t)row * NHEADS + d] = kwraw[(size_t)row * KW_N + HDIM + d] * W_SCALE;
}

// ---------------------------------------------------------------------------
// q GEMM + fused rope + scale: qr_bf16(4096x1536) @ wq_b^T(8192x1536)^T
// wave tile 16x64 aligned so rope pairs (d, d+32) live in tiles (t, t+2).
// ---------------------------------------------------------------------------
__global__ __launch_bounds__(32) void gemm_q_rope_kernel(
    const __bf16* __restrict__ A, const __bf16* __restrict__ BT,
    const int* __restrict__ pos, __bf16* __restrict__ qout) {
  const int lane = threadIdx.x;
  const int n0 = blockIdx.x * 64, i0 = blockIdx.y * 16;
  v8f acc[4];
#pragma unroll
  for (int t = 0; t < 4; ++t) acc[t] = v8f_zero();

  for (int k0 = 0; k0 < QLR; k0 += 32) {
    v16bf a = load_tile16x32(A, QLR, i0, k0, lane);
#pragma unroll
    for (int t = 0; t < 4; ++t) {
      v16bf b = load_tile16x32(BT, QLR, n0 + t * 16, k0, lane);
      acc[t] = wmma_bf16(a, b, acc[t]);
    }
  }

  const int half8 = (lane & 16) ? 8 : 0;
  const int ln = lane & 15;
  const int chunk = (n0 >> 6) & 1;  // 0: rope dims 0..63 of head, 1: passthrough

  if (chunk == 0) {
#pragma unroll
    for (int v = 0; v < 8; ++v) {
      const int m = i0 + v + half8;
      const int p = pos[m];
#pragma unroll
      for (int t = 0; t < 2; ++t) {
        const int d = t * 16 + ln;              // 0..31 (rope half index)
        float inv = __expf((float)d * NEG_LOG_BASE_OVER_HALF);
        float ang = (float)p * inv;
        float s_ = __sinf(ang), c_ = __cosf(ang);
        float x1 = acc[t][v], x2 = acc[t + 2][v];
        qout[(size_t)m * QCOLS + n0 + t * 16 + ln] =
            (__bf16)((x1 * c_ - x2 * s_) * Q_SCALE);
        qout[(size_t)m * QCOLS + n0 + (t + 2) * 16 + ln] =
            (__bf16)((x2 * c_ + x1 * s_) * Q_SCALE);
      }
    }
  } else {
#pragma unroll
    for (int t = 0; t < 4; ++t)
#pragma unroll
      for (int v = 0; v < 8; ++v)
        qout[(size_t)(i0 + v + half8) * QCOLS + n0 + t * 16 + ln] =
            (__bf16)(acc[t][v] * Q_SCALE);
  }
}

// ---------------------------------------------------------------------------
// fill scores with -inf (for the fully-masked tiles we skip)
// ---------------------------------------------------------------------------
__global__ void fill_neg_inf(float* __restrict__ p, int n) {
  int i = blockIdx.x * blockDim.x + threadIdx.x;
  if (i < n) p[i] = -__builtin_inff();
}

// ---------------------------------------------------------------------------
// stage one head's 16xHDIM q strip (4KB bf16) into LDS; 256 threads,
// 16 elements (2x b128) per thread. Async path uses CDNA5 ASYNCcnt DMA.
// ---------------------------------------------------------------------------
__device__ __forceinline__ void stage_q_strip(const __bf16* __restrict__ q,
                                              __bf16* dst, int i0, int h, int tid) {
  const int idx = tid << 4;        // 16 elems/thread, 256 threads = 32*128
  const int row = idx >> 7;        // / HDIM
  const int col = idx & (HDIM - 1);
  const __bf16* src = q + (size_t)(i0 + row) * QCOLS + h * HDIM + col;
#if USE_ASYNC_LDS
  __builtin_amdgcn_global_load_async_to_lds_b128(
      (__attribute__((address_space(1))) v4i_vec*)(src),
      (__attribute__((address_space(3))) v4i_vec*)(dst + idx), 0, 0);
  __builtin_amdgcn_global_load_async_to_lds_b128(
      (__attribute__((address_space(1))) v4i_vec*)(src + 8),
      (__attribute__((address_space(3))) v4i_vec*)(dst + idx + 8), 0, 0);
#else
  *(v8bf*)(dst + idx)     = *(const v8bf*)(src);
  *(v8bf*)(dst + idx + 8) = *(const v8bf*)(src + 8);
#endif
}

__device__ __forceinline__ void wait_async_stage() {
#if USE_ASYNC_LDS
  __builtin_amdgcn_s_wait_asynccnt(0);
#endif
}

// ---------------------------------------------------------------------------
// scores: 8-wave workgroup shares one 16-row q tile. Per head, the 4KB q strip
// is staged into LDS (double-buffered, async DMA) once per block instead of
// once per wave. Each wave owns a 16x32 score tile; its k-tiles (B operands,
// shared across all 64 heads) are hoisted into 64 VGPRs.
// ---------------------------------------------------------------------------
__global__ __launch_bounds__(256) void scores_kernel(
    const __bf16* __restrict__ q, const __bf16* __restrict__ k,
    const float* __restrict__ w, float* __restrict__ sc) {
  const int tid  = threadIdx.x;
  const int lane = tid & 31;
  const int wv   = tid >> 5;                 // wave index 0..7
  const int i0 = blockIdx.y * 16;
  const int jbase = blockIdx.x * 256;
  if (jbase > i0 + 15) return;               // whole block causally masked
  const int j0 = jbase + wv * 32;
  const bool active = (j0 <= i0 + 15);       // wave-uniform -> EXEC stays all-1

  __shared__ __bf16 qsm[2][16 * HDIM];       // double-buffered q strip, 8KB

  // hoist B: k rows j0..j0+31, K = 128 -> 8 tiles of 16x32 (64 VGPRs)
  v16bf bt[8];
  if (active) {
#pragma unroll
    for (int nt = 0; nt < 2; ++nt)
#pragma unroll
      for (int kk = 0; kk < 4; ++kk)
        bt[nt * 4 + kk] = load_tile16x32(k, HDIM, j0 + nt * 16, kk * 32, lane);
  }

  v8f f0 = v8f_zero(), f1 = v8f_zero();
  const int half8 = (lane & 16) ? 8 : 0;
  const int ln = lane & 15;

  stage_q_strip(q, qsm[0], i0, 0, tid);
  wait_async_stage();
  __syncthreads();

  for (int h = 0; h < NHEADS; ++h) {
    const __bf16* cur = qsm[h & 1];
    if (h + 1 < NHEADS)                       // prefetch next head's strip
      stage_q_strip(q, qsm[(h + 1) & 1], i0, h + 1, tid);

    if (active) {
      v8f p0 = v8f_zero(), p1 = v8f_zero();
#pragma unroll
      for (int kk = 0; kk < 4; ++kk) {
        v16bf a = load_tile_lds(cur, kk * 32, lane);
        p0 = wmma_bf16(a, bt[kk],     p0);
        p1 = wmma_bf16(a, bt[4 + kk], p1);
      }
      float wl = w[(size_t)(i0 + ln) * NHEADS + h];  // lane L holds w[row i0+L%16]
#pragma unroll
      for (int v = 0; v < 8; ++v) {
        float wv_ = __shfl(wl, v + half8, 32);       // w for this VGPR's row
        f0[v] += wv_ * fmaxf(p0[v], 0.f);
        f1[v] += wv_ * fmaxf(p1[v], 0.f);
      }
    }
    if (h + 1 < NHEADS) {
      wait_async_stage();
      __syncthreads();
    }
  }

  if (active) {
#pragma unroll
    for (int v = 0; v < 8; ++v) {
      const int m = i0 + v + half8;
      const int n = j0 + ln;
      sc[(size_t)m * T_TOK + n]      = (n      <= m) ? f0[v] : -__builtin_inff();
      sc[(size_t)m * T_TOK + n + 16] = (n + 16 <= m) ? f1[v] : -__builtin_inff();
    }
  }
}

// ---------------------------------------------------------------------------
// per-row top-2048 via full bitonic sort of 4096 keys in LDS.
// key = (~sortable(val) << 32) | idx  -> ascending sort = desc value, asc idx.
// ---------------------------------------------------------------------------
__global__ __launch_bounds__(256) void topk_kernel(const float* __restrict__ sc,
                                                   int* __restrict__ out) {
  __shared__ unsigned long long keys[T_TOK];
  const int row = blockIdx.x, tid = threadIdx.x;

  for (int j = tid; j < T_TOK; j += 256) {
    unsigned u = __float_as_uint(sc[(size_t)row * T_TOK + j]);
    u = (u & 0x80000000u) ? ~u : (u | 0x80000000u);  // order-preserving map
    keys[j] = ((unsigned long long)(~u) << 32) | (unsigned)j;
  }
  __syncthreads();

  for (int kk = 2; kk <= T_TOK; kk <<= 1) {
    for (int j = kk >> 1; j > 0; j >>= 1) {
      for (int i = tid; i < T_TOK; i += 256) {
        int ixj = i ^ j;
        if (ixj > i) {
          bool asc = ((i & kk) == 0);
          unsigned long long a = keys[i], b = keys[ixj];
          if ((a > b) == asc) { keys[i] = b; keys[ixj] = a; }
        }
      }
      __syncthreads();
    }
  }
  for (int t = tid; t < TOPK; t += 256)
    out[(size_t)row * TOPK + t] = (int)(keys[t] & 0xFFFFFFFFull);
}

// ---------------------------------------------------------------------------
extern "C" void kernel_launch(void* const* d_in, const int* in_sizes, int n_in,
                              void* d_out, int out_size, void* d_ws, size_t ws_size,
                              hipStream_t stream) {
  const float* hidden = (const float*)d_in[0];  // (4096, 7168)
  const float* qr     = (const float*)d_in[1];  // (4096, 1536)
  const int*   pos    = (const int*)d_in[2];    // (4096,)
  const float* wq_b   = (const float*)d_in[3];  // (1536, 8192)
  const float* wk     = (const float*)d_in[4];  // (7168, 128)
  const float* kgam   = (const float*)d_in[5];  // (128,)
  const float* kbet   = (const float*)d_in[6];  // (128,)
  const float* wproj  = (const float*)d_in[7];  // (7168, 64)

  char* ws = (char*)d_ws;
  __bf16* qr_bf  = (__bf16*)(ws + OFF_QRB);
  __bf16* wqbt   = (__bf16*)(ws + OFF_WQBT);
  __bf16* hid_bf = (__bf16*)(ws + OFF_HIDB);
  __bf16* wkt    = (__bf16*)(ws + OFF_WKT);
  float*  kwraw  = (float*)(ws + OFF_KWR);
  __bf16* q_bf   = (__bf16*)(ws + OFF_QB);
  __bf16* k_bf   = (__bf16*)(ws + OFF_KB);
  float*  w_f    = (float*)(ws + OFF_W);

  int*   topk = (int*)d_out;                                  // (4096, 2048)
  float* sc   = (float*)d_out + (size_t)T_TOK * TOPK;         // (4096, 4096)

  // 1) precision conversion
  {
    int n = T_TOK * QLR;
    cvt_f32_bf16<<<(n + 255) / 256, 256, 0, stream>>>(qr, qr_bf, n);
  }
  {
    int n = T_TOK * H_SIZE;
    cvt_f32_bf16<<<(n + 255) / 256, 256, 0, stream>>>(hidden, hid_bf, n);
  }
  // 2) weight transposes (K-contiguous B operands for b128 tile loads)
  transpose_cvt<<<dim3(QCOLS / 32, QLR / 32), dim3(32, 8), 0, stream>>>(
      wq_b, wqbt, QLR, QCOLS, QLR);
  transpose_cvt<<<dim3(HDIM / 32, H_SIZE / 32), dim3(32, 8), 0, stream>>>(
      wk, wkt, H_SIZE, HDIM, H_SIZE);
  transpose_cvt<<<dim3(NHEADS / 32, H_SIZE / 32), dim3(32, 8), 0, stream>>>(
      wproj, wkt + (size_t)HDIM * H_SIZE, H_SIZE, NHEADS, H_SIZE);

  // 3) k/w projection (WMMA), 4) layernorm + rope + w scale
  gemm_kw_kernel<<<dim3(KW_N / 64, T_TOK / 16), 32, 0, stream>>>(hid_bf, wkt, kwraw);
  ln_rope_kw<<<T_TOK, 128, 0, stream>>>(kwraw, kgam, kbet, pos, k_bf, w_f);

  // 5) q projection + fused rope (WMMA)
  gemm_q_rope_kernel<<<dim3(QCOLS / 64, T_TOK / 16), 32, 0, stream>>>(
      qr_bf, wqbt, pos, q_bf);

  // 6) scores: -inf init then causal WMMA accumulation over heads
  {
    int n = T_TOK * T_TOK;
    fill_neg_inf<<<(n + 255) / 256, 256, 0, stream>>>(sc, n);
  }
  scores_kernel<<<dim3(T_TOK / 256, T_TOK / 16), 256, 0, stream>>>(q_bf, k_bf, w_f, sc);

  // 7) per-row top-2048
  topk_kernel<<<T_TOK, 256, 0, stream>>>(sc, topk);
}
